// PowerCoderAttention_14328010899463
// MI455X (gfx1250) — matrix operand order repaired and checked by reading, not compiled, not run on hardware
//
#include <hip/hip_runtime.h>

// ---------------------------------------------------------------------------
// PowerCoderAttention for MI455X (gfx1250, wave32, WMMA bf16)
// - B operands staged K-contiguous (fragments = 2x ds_load_b128)
// - tile staging via GLOBAL_LOAD_ASYNC_TO_LDS (ASYNCcnt), double-buffered so
//   the async copy of tile t+1 overlaps the WMMA burst on tile t.
// ---------------------------------------------------------------------------
typedef __bf16 bf16;
typedef __attribute__((ext_vector_type(16))) __bf16 v16bf;
typedef __attribute__((ext_vector_type(8)))  float  v8f;
typedef int vi4 __attribute__((vector_size(16)));
typedef int vi2 __attribute__((vector_size(8)));

#define S_LEN 2048
#define HIDN  2048
#define NH    16
#define NKV   4
#define DH    128
#define NQKV  3072      // 2048 (Q) + 512 (K) + 512 (V)
#define QK_SCALE 0.08838834764831845f   // 128^-0.5

#if __has_builtin(__builtin_amdgcn_global_load_async_to_lds_b128) && \
    __has_builtin(__builtin_amdgcn_global_load_async_to_lds_b64)  && \
    __has_builtin(__builtin_amdgcn_s_wait_asynccnt)
#define HAVE_ASYNC_LDS 1
#endif

#define AS_GLOBAL __attribute__((address_space(1)))
#define AS_SHARED __attribute__((address_space(3)))

static __device__ __forceinline__ void stage_b128(void* lds, const void* gsrc) {
#ifdef HAVE_ASYNC_LDS
  __builtin_amdgcn_global_load_async_to_lds_b128(
      (AS_GLOBAL vi4*)((void*)gsrc),
      (AS_SHARED vi4*)lds, 0, 0);
#else
  *(uint4*)lds = *(const uint4*)gsrc;
#endif
}

static __device__ __forceinline__ void stage_b64(void* lds, const void* gsrc) {
#ifdef HAVE_ASYNC_LDS
  __builtin_amdgcn_global_load_async_to_lds_b64(
      (AS_GLOBAL vi2*)((void*)gsrc),
      (AS_SHARED vi2*)lds, 0, 0);
#else
  *(uint2*)lds = *(const uint2*)gsrc;
#endif
}

static __device__ __forceinline__ void stage_wait() {
#ifdef HAVE_ASYNC_LDS
  __builtin_amdgcn_s_wait_asynccnt(0);
#endif
}

static __device__ __forceinline__ v8f wmma_bf16(v16bf a, v16bf b, v8f c) {
  // D = A(16x32 bf16) * B(32x16 bf16) + C(16x16 f32)
  return __builtin_amdgcn_wmma_f32_16x16x32_bf16(false, a, false, b,
                                                 (short)0, c, false, false);
}

union AFragU { v16bf v; uint4 q[2]; };

// A fragment (16x32 tile, row-major [m][k] in LDS, stride `ld` elems;
// ld must be a multiple of 8 so 16B alignment holds).
// lane<16: row=lane, K 0..7 & 16..23 ; lane>=16: row=lane-16, K 8..15 & 24..31
static __device__ __forceinline__ v16bf load_a_frag(const bf16* base, int ld, int lane) {
  const int row = lane & 15;
  const int ka  = (lane >> 4) * 8;
  const bf16* p = base + row * ld + ka;
  AFragU f;
  f.q[0] = *(const uint4*)(p);
  f.q[1] = *(const uint4*)(p + 16);
  return f.v;
}

// B fragment from a K-contiguous ("transposed") tile: LDS layout [n][k],
// stride `ld` elems (multiple of 8). lane: col n = lane&15, K kb..kb+7 &
// kb+16..kb+23 where kb=(lane>>4)*8 -> 2x b128 (matches A fragment order).
static __device__ __forceinline__ v16bf load_bt_frag(const bf16* base, int ld, int lane) {
  const int n  = lane & 15;
  const int kb = (lane >> 4) * 8;
  const bf16* p = base + n * ld + kb;
  AFragU f;
  f.q[0] = *(const uint4*)(p);
  f.q[1] = *(const uint4*)(p + 16);
  return f.v;
}

// ---------------------------------------------------------------------------
// Converts / packing (all weight matrices are written TRANSPOSED: [n][k])
// ---------------------------------------------------------------------------
__global__ __launch_bounds__(256) void cvt_f32_bf16(const float* __restrict__ src,
                                                    bf16* __restrict__ dst, int n) {
  for (int i = blockIdx.x * 256 + threadIdx.x; i < n; i += gridDim.x * 256)
    dst[i] = (bf16)src[i];
}

// Wqkv^T : dst[n][c], n in [0,3072), c in [0,2048)
__global__ __launch_bounds__(256) void pack_wqkv_t(const float* __restrict__ Wq,
                                                   const float* __restrict__ Wk,
                                                   const float* __restrict__ Wv,
                                                   bf16* __restrict__ dst) {
  for (int i = blockIdx.x * 256 + threadIdx.x; i < NQKV * HIDN; i += gridDim.x * 256) {
    const int n = i >> 11;          // / HIDN
    const int c = i & (HIDN - 1);
    float v;
    if (n < 2048)      v = Wq[c * 2048 + n];
    else if (n < 2560) v = Wk[c * 512 + (n - 2048)];
    else               v = Wv[c * 512 + (n - 2560)];
    dst[i] = (bf16)v;
  }
}

// Wo^T : dst[n][k] = Wo[k][n], both 2048x2048
__global__ __launch_bounds__(256) void cvt_wo_t(const float* __restrict__ Wo,
                                                bf16* __restrict__ dst) {
  for (int i = blockIdx.x * 256 + threadIdx.x; i < HIDN * HIDN; i += gridDim.x * 256) {
    const int n = i >> 11;
    const int k = i & (HIDN - 1);
    dst[i] = (bf16)Wo[(k << 11) + n];
  }
}

// ---------------------------------------------------------------------------
// Gate: g[kv][s] = log_sigmoid(hidden[s] . Wg[:,kv]); G = cumsum_s(g)
// ---------------------------------------------------------------------------
__global__ __launch_bounds__(256) void gate_kernel(const float* __restrict__ hidden,
                                                   const float* __restrict__ Wg,
                                                   float* __restrict__ G) {
  const int kv = blockIdx.x;
  const int t  = threadIdx.x;
  __shared__ float wg[HIDN];
  __shared__ float sums[256];
  for (int c = t; c < HIDN; c += 256) wg[c] = Wg[c * NKV + kv];
  __syncthreads();

  float loc[8];
  float run = 0.f;
#pragma unroll 1
  for (int i = 0; i < 8; ++i) {
    const int s = t * 8 + i;
    const float* hrow = hidden + (size_t)s * HIDN;
    float acc = 0.f;
    for (int c = 0; c < HIDN; ++c) acc = fmaf(hrow[c], wg[c], acc);
    // log_sigmoid(x) = min(x,0) - log1p(exp(-|x|))
    const float ls = fminf(acc, 0.f) - log1pf(__expf(-fabsf(acc)));
    run += ls;
    loc[i] = run;
  }
  sums[t] = run;
  __syncthreads();
  for (int off = 1; off < 256; off <<= 1) {
    float v = (t >= off) ? sums[t - off] : 0.f;
    __syncthreads();
    sums[t] += v;
    __syncthreads();
  }
  const float base = (t > 0) ? sums[t - 1] : 0.f;
#pragma unroll
  for (int i = 0; i < 8; ++i) G[kv * S_LEN + t * 8 + i] = base + loc[i];
}

// ---------------------------------------------------------------------------
// bf16 GEMM, f32 out: C[M][N] = A[M][K] * Bt[N][K]^T   (B passed TRANSPOSED)
// block tile 128x64, 8 waves = 4(M) x 2(N), each wave 32x32 (2x2 WMMA tiles)
// Double-buffered async LDS staging: copy of K-slab t+1 overlaps WMMA on t.
// ---------------------------------------------------------------------------
__global__ __launch_bounds__(256) void gemm_bf16_f32(const bf16* __restrict__ A,
                                                     const bf16* __restrict__ Bt,
                                                     float* __restrict__ C,
                                                     int M, int N, int K) {
  __shared__ __align__(16) bf16 As[2][128][40];  // [m][k], 32 K used
  __shared__ __align__(16) bf16 Bs[2][64][40];   // [n][k], 32 K used
  const int lane = threadIdx.x & 31;
  const int wave = threadIdx.x >> 5;
  const int wm = wave >> 1;      // 0..3
  const int wn = wave & 1;       // 0..1
  const size_t tM = (size_t)blockIdx.x * 128;
  const size_t tN = (size_t)blockIdx.y * 64;

  const int ar  = threadIdx.x >> 1;          // A: 128 rows x 2 segs of 16
  const int asg = (threadIdx.x & 1) * 16;
  const int bn  = threadIdx.x >> 2;          // B: 64 rows x 4 segs of 8
  const int bsg = (threadIdx.x & 3) * 8;

  v8f acc00 = {}, acc01 = {}, acc10 = {}, acc11 = {};

  // prologue: stage K-slab 0 into buffer 0
  stage_b128(&As[0][ar][asg], A  + (tM + ar) * K + asg);
  stage_b64 (&Bs[0][bn][bsg], Bt + (tN + bn) * K + bsg);

  const int nK = K >> 5;
  for (int it = 0; it < nK; ++it) {
    const int cur = it & 1;
    stage_wait();
    __syncthreads();
    if (it + 1 < nK) {
      const int k0 = (it + 1) << 5;
      stage_b128(&As[cur ^ 1][ar][asg], A  + (tM + ar) * K + k0 + asg);
      stage_b64 (&Bs[cur ^ 1][bn][bsg], Bt + (tN + bn) * K + k0 + bsg);
    }
    const v16bf a0 = load_a_frag (&As[cur][wm * 32][0],      40, lane);
    const v16bf a1 = load_a_frag (&As[cur][wm * 32 + 16][0], 40, lane);
    const v16bf b0 = load_bt_frag(&Bs[cur][wn * 32][0],      40, lane);
    const v16bf b1 = load_bt_frag(&Bs[cur][wn * 32 + 16][0], 40, lane);
    acc00 = wmma_bf16(a0, b0, acc00);
    acc01 = wmma_bf16(a0, b1, acc01);
    acc10 = wmma_bf16(a1, b0, acc10);
    acc11 = wmma_bf16(a1, b1, acc11);
  }

  const int cr = (lane >> 4) * 8;
  const int cc = lane & 15;
#pragma unroll
  for (int r = 0; r < 8; ++r) {
    const size_t row0 = tM + wm * 32 + cr + r;
    const size_t row1 = row0 + 16;
    C[row0 * N + tN + wn * 32 + cc]      = acc00[r];
    C[row0 * N + tN + wn * 32 + 16 + cc] = acc01[r];
    C[row1 * N + tN + wn * 32 + cc]      = acc10[r];
    C[row1 * N + tN + wn * 32 + 16 + cc] = acc11[r];
  }
}

// ---------------------------------------------------------------------------
// RoPE + repack: raw f32 [s][3072] -> Qb bf16 [h][s][d],
//                Kb bf16 [kv][s][d] (natural = "B transposed" for qk),
//                Vt bf16 [kv][d][s] (transposed = "B transposed" for PV)
// ---------------------------------------------------------------------------
__global__ __launch_bounds__(256) void rope_repack(const float* __restrict__ raw,
                                                   const float* __restrict__ cosb,
                                                   const float* __restrict__ sinb,
                                                   bf16* __restrict__ Qb,
                                                   bf16* __restrict__ Kb,
                                                   bf16* __restrict__ Vt) {
  const int s = blockIdx.x;
  const float* r = raw + (size_t)s * NQKV;
  __shared__ float cs[DH], sn[DH];
  if (threadIdx.x < DH) {
    cs[threadIdx.x] = cosb[s * DH + threadIdx.x];
    sn[threadIdx.x] = sinb[s * DH + threadIdx.x];
  }
  __syncthreads();
  for (int n = threadIdx.x; n < NH * DH; n += 256) {
    const int h = n >> 7, d = n & 127;
    const float v = r[n];
    const float p = r[(h << 7) | (d ^ 64)];
    const float rot = (d < 64) ? -p : p;
    Qb[((size_t)h * S_LEN + s) * DH + d] = (bf16)(v * cs[d] + rot * sn[d]);
  }
  for (int n = threadIdx.x; n < NKV * DH; n += 256) {
    const int kv = n >> 7, d = n & 127;
    const float v = r[2048 + n];
    const float p = r[2048 + ((kv << 7) | (d ^ 64))];
    const float rot = (d < 64) ? -p : p;
    Kb[((size_t)kv * S_LEN + s) * DH + d] = (bf16)(v * cs[d] + rot * sn[d]);
  }
  for (int n = threadIdx.x; n < NKV * DH; n += 256) {
    const int kv = n >> 7, d = n & 127;
    Vt[((size_t)(kv * DH + d)) * S_LEN + s] = (bf16)r[2560 + n];
  }
}

// ---------------------------------------------------------------------------
// Fused decay attention.
// grid = (S/128, NH). Block = 256 threads = 8 waves; wave w owns query rows
// [qt + 16w, qt + 16w + 16) and the full D=128 output for one head.
// Double-buffered async staging of 32-key K/V tiles.
// ---------------------------------------------------------------------------
__global__ __launch_bounds__(256) void attn_kernel(const bf16* __restrict__ Qb,
                                                   const bf16* __restrict__ Kb,
                                                   const bf16* __restrict__ Vt,
                                                   const float* __restrict__ G,
                                                   bf16* __restrict__ O) {
  const int h    = blockIdx.y;
  const int kv   = h >> 2;                // GROUPS = 4
  const int qt   = blockIdx.x * 128;
  const int lane = threadIdx.x & 31;
  const int wave = threadIdx.x >> 5;

  // LDS layout (48128 B):
  //   [0,     10240) : Ss  8 waves * 16x40 score staging (loop)
  //   [10240, 48128) double buffer b=0,1 at 10240 + b*18944:
  //       Kls_b 32x136 [key][d]  (8704 B)
  //       Vls_b 128x40 [d][key] (10240 B)
  //   Qs 128x136 (34816 B) overlaps everything; used only in phase 0.
  __shared__ __align__(16) char smem[48128];
  bf16 (*Qs)[136] = (bf16(*)[136])smem;
  bf16 (*Ss)[40]  = (bf16(*)[40])smem;
  char* kvbase = smem + 10240;
#define KLS(b) ((bf16(*)[136])(kvbase + (b) * 18944))
#define VLS(b) ((bf16(*)[40]) (kvbase + (b) * 18944 + 8704))

  // ---- phase 0: stage Q tile (128x128) and pull per-wave A fragments ----
#pragma unroll
  for (int it = 0; it < 4; ++it) {
    const int u   = threadIdx.x + it * 256;   // 1024 units of 16 elems
    const int row = u >> 3;
    const int sg  = (u & 7) * 16;
    stage_b128(&Qs[row][sg], Qb + ((size_t)h * S_LEN + qt + row) * DH + sg);
  }
  stage_wait();
  __syncthreads();
  v16bf aq[4];
#pragma unroll
  for (int c = 0; c < 4; ++c)
    aq[c] = load_a_frag(&Qs[wave * 16][c * 32], 136, lane);
  __syncthreads();   // Q gathers done -> loop buffers may overwrite Qs

  // per-lane query-row decay state G_i (rows (lane>>4)*8 + r of this wave)
  float Gi[8];
#pragma unroll
  for (int r = 0; r < 8; ++r)
    Gi[r] = G[kv * S_LEN + qt + wave * 16 + (lane >> 4) * 8 + r];

  const int krow = threadIdx.x >> 3;          // K stage: key 0..31
  const int ksg  = (threadIdx.x & 7) * 16;    //          d segment
  const int vrow = threadIdx.x >> 1;          // V stage: d 0..127
  const int vsg  = (threadIdx.x & 1) * 16;    //          key segment

  v8f  oacc[8] = {};       // 8 x 16-wide d sub-tiles
  float rs[8]  = {0.f, 0.f, 0.f, 0.f, 0.f, 0.f, 0.f, 0.f};

  // prologue: stage key tile 0 into buffer 0
  stage_b128(&KLS(0)[krow][ksg], Kb + ((size_t)kv * S_LEN + krow) * DH + ksg);
  stage_b128(&VLS(0)[vrow][vsg], Vt + ((size_t)(kv * DH + vrow)) * S_LEN + vsg);

  const int nIter = (qt + 128) >> 5;
  for (int it = 0; it < nIter; ++it) {
    const int cur = it & 1;
    const int jt  = it << 5;
    stage_wait();
    __syncthreads();
    if (it + 1 < nIter) {
      const int jn = (it + 1) << 5;
      stage_b128(&KLS(cur ^ 1)[krow][ksg],
                 Kb + ((size_t)kv * S_LEN + jn + krow) * DH + ksg);
      stage_b128(&VLS(cur ^ 1)[vrow][vsg],
                 Vt + ((size_t)(kv * DH + vrow)) * S_LEN + jn + vsg);
    }
    bf16 (*Kc)[136] = KLS(cur);
    bf16 (*Vc)[40]  = VLS(cur);

    // ---- qk for this wave's 16 rows x 32 keys, then decay/mask/square ----
#pragma unroll
    for (int tk = 0; tk < 2; ++tk) {
      v8f qk = {};
#pragma unroll
      for (int c = 0; c < 4; ++c) {
        const v16bf bk = load_bt_frag(&Kc[tk * 16][c * 32], 136, lane);
        qk = wmma_bf16(aq[c], bk, qk);
      }
      const int j  = jt + tk * 16 + (lane & 15);
      const float Gj = G[kv * S_LEN + j];
#pragma unroll
      for (int r = 0; r < 8; ++r) {
        const int i = qt + wave * 16 + (lane >> 4) * 8 + r;
        const float v = qk[r] * QK_SCALE;
        const float sc = (j <= i) ? v * v * __expf(Gi[r] - Gj) : 0.f;
        rs[r] += sc;
        Ss[wave * 16 + (lane >> 4) * 8 + r][tk * 16 + (lane & 15)] = (bf16)sc;
      }
    }
    // wave-private LDS patch: DS ops are in-order within a wave -> no barrier
    const v16bf asc = load_a_frag(&Ss[wave * 16][0], 40, lane);
#pragma unroll
    for (int nt = 0; nt < 8; ++nt) {
      const v16bf bv = load_bt_frag(&Vc[nt * 16][0], 40, lane);
      oacc[nt] = wmma_bf16(asc, bv, oacc[nt]);
    }
  }

  // ---- denominator: reduce row sums across the 16 lanes of each half-wave ----
#pragma unroll
  for (int r = 0; r < 8; ++r) {
    float v = rs[r];
#pragma unroll
    for (int m = 1; m < 16; m <<= 1) v += __shfl_xor(v, m, 32);
    rs[r] = fmaxf(v, 1.0f);
  }

  // ---- write attn output bf16 [s][h*128 + d] ----
  const int cr = (lane >> 4) * 8;
  const int cc = lane & 15;
#pragma unroll
  for (int nt = 0; nt < 8; ++nt) {
#pragma unroll
    for (int r = 0; r < 8; ++r) {
      const size_t i = qt + wave * 16 + cr + r;
      O[i * (size_t)HIDN + h * DH + nt * 16 + cc] = (bf16)(oacc[nt][r] / rs[r]);
    }
  }
#undef KLS
#undef VLS
}

// ---------------------------------------------------------------------------
// Launch
// ---------------------------------------------------------------------------
extern "C" void kernel_launch(void* const* d_in, const int* in_sizes, int n_in,
                              void* d_out, int out_size, void* d_ws, size_t ws_size,
                              hipStream_t stream) {
  const float* hidden = (const float*)d_in[0];
  const float* cosb   = (const float*)d_in[1];
  const float* sinb   = (const float*)d_in[2];
  const float* Wq     = (const float*)d_in[3];
  const float* Wk     = (const float*)d_in[4];
  const float* Wv     = (const float*)d_in[5];
  const float* Wg     = (const float*)d_in[6];
  const float* Wo     = (const float*)d_in[7];
  float* out = (float*)d_out;
  char*  ws  = (char*)d_ws;

  // Workspace layout (regions aliased across pipeline phases):
  //  A [0, 8.4MB)      : hidden bf16        -> later Kb (2MB) + Vt (2MB)
  //  B [8.4, 21.0MB)   : Wqkv^T bf16        -> later Qb (8.4MB)
  //  C [21.0, 29.4MB)  : Wo^T bf16 (persists)
  //  D [29.4, 54.5MB)  : QKV raw f32        -> later attn out bf16 (8.4MB)
  //  E [54.5MB, +32KB) : G f32
  const size_t OFF_A = 0;
  const size_t OFF_B = 8388608;
  const size_t OFF_C = 20971520;
  const size_t OFF_D = 29360128;
  const size_t OFF_E = 54525952;

  bf16*  hb     = (bf16*)(ws + OFF_A);
  bf16*  wqkvt  = (bf16*)(ws + OFF_B);
  bf16*  wot    = (bf16*)(ws + OFF_C);
  float* qkvraw = (float*)(ws + OFF_D);
  float* Gbuf   = (float*)(ws + OFF_E);
  bf16*  Qb     = (bf16*)(ws + OFF_B);             // over dead Wqkv^T
  bf16*  Kb     = (bf16*)(ws + OFF_A);             // over dead hidden-bf16
  bf16*  Vt     = (bf16*)(ws + OFF_A + 2097152);
  bf16*  attn   = (bf16*)(ws + OFF_D);             // over dead QKV raw

  cvt_f32_bf16<<<2048, 256, 0, stream>>>(hidden, hb, S_LEN * HIDN);
  pack_wqkv_t <<<2048, 256, 0, stream>>>(Wq, Wk, Wv, wqkvt);
  cvt_wo_t    <<<2048, 256, 0, stream>>>(Wo, wot);
  gate_kernel <<<NKV,  256, 0, stream>>>(hidden, Wg, Gbuf);

  gemm_bf16_f32<<<dim3(S_LEN / 128, NQKV / 64), 256, 0, stream>>>(
      hb, wqkvt, qkvraw, S_LEN, NQKV, HIDN);

  rope_repack<<<S_LEN, 256, 0, stream>>>(qkvraw, cosb, sinb, Qb, Kb, Vt);

  attn_kernel<<<dim3(S_LEN / 128, NH), 256, 0, stream>>>(Qb, Kb, Vt, Gbuf, attn);

  gemm_bf16_f32<<<dim3(S_LEN / 128, HIDN / 64), 256, 0, stream>>>(
      attn, wot, out, S_LEN, HIDN, HIDN);
}
